// MultiHeadAttention_58411555226232
// MI455X (gfx1250) — matrix operand order repaired and checked by reading, not compiled, not run on hardware
//
#include <hip/hip_runtime.h>
#include <hip/hip_bf16.h>
#include <stdint.h>

// Problem constants (match reference)
#define DM    1024
#define SEQ   2048
#define NHEAD 16
#define DK    64
#define BSZ   2
#define SCALE 0.125f          // 1/sqrt(64)
#define NEGBIG -1.0e10f

typedef __attribute__((ext_vector_type(4)))  float        f32x4;
typedef __attribute__((ext_vector_type(8)))  float        f32x8;
typedef __attribute__((ext_vector_type(8)))  __bf16       v8bf;
typedef __attribute__((ext_vector_type(16))) __bf16       v16bf;
typedef __attribute__((ext_vector_type(8)))  float        v8f;
typedef __attribute__((ext_vector_type(4)))  unsigned int u32x4;
typedef __attribute__((ext_vector_type(8)))  int          i32x8;
typedef __attribute__((ext_vector_type(4)))  int          i32x4;

// ---------------------------------------------------------------------------
// WMMA fragment helpers (layouts per CDNA5 ISA 7.12.2, bf16 16x16x32)
// A (16xK): lane l<16 -> row l, K {k0+0..7, k0+16..23}; lane l>=16 -> row l-16,
//           K {k0+8..15, k0+24..31}
// B (Kx16): lane n<16 -> col n, K {k0+0..7, k0+16..23}; lane n>=16 -> col n-16,
//           K {k0+8..15, k0+24..31}
// Sources are row-major with K contiguous (B source holds B^T: Src[n][k]=B[k][n]).
// ---------------------------------------------------------------------------
__device__ inline v16bf make_v16bf(f32x4 lo, f32x4 hi) {
  f32x8 t;
  t[0] = lo[0]; t[1] = lo[1]; t[2] = lo[2]; t[3] = lo[3];
  t[4] = hi[0]; t[5] = hi[1]; t[6] = hi[2]; t[7] = hi[3];
  return __builtin_bit_cast(v16bf, t);
}

__device__ inline v16bf frag_a16(const __bf16* m, int ld, int k0) {
  const int lane = threadIdx.x & 31;
  const __bf16* p = m + (lane & 15) * ld + k0 + ((lane >> 4) << 3);
  f32x4 lo = *(const f32x4*)p;         // k +0..7
  f32x4 hi = *(const f32x4*)(p + 16);  // k +16..23
  return make_v16bf(lo, hi);
}

__device__ inline v16bf frag_b16(const __bf16* m, int ld, int k0) {
  const int lane = threadIdx.x & 31;
  const __bf16* p = m + (lane & 15) * ld + k0 + ((lane >> 4) << 3);
  f32x4 lo = *(const f32x4*)p;         // k +0..7
  f32x4 hi = *(const f32x4*)(p + 16);  // k +16..23
  return make_v16bf(lo, hi);
}

__device__ inline v8f wmma_bf16(v16bf a, v16bf b, v8f c) {
  return __builtin_amdgcn_wmma_f32_16x16x32_bf16(
      /*neg_a=*/false, a, /*neg_b=*/false, b,
      /*c_mod=*/(short)0, c, /*reuse_a=*/false, /*reuse_b=*/false);
}

// ---------------------------------------------------------------------------
// Tensor Data Mover: 2D bf16 tile (tileK x tileRows) -> LDS, contiguous rows.
// D# built per CDNA5 ISA §8 (group0: count/lds/global/type, group1: dims).
// Wave-level op (EXEC ignored); issue from one wave, wait TENSORcnt, barrier.
// This toolchain exposes the 6-arg builtin (4 descriptor groups + spare + cpol).
// ---------------------------------------------------------------------------
__device__ inline void tdm_load_2d_bf16(unsigned ldsOff, const void* gptr,
                                        int tileK, int tileRows,
                                        int rowStrideElems, int tensorRows) {
  const unsigned long long ga = (unsigned long long)(uintptr_t)gptr;
  u32x4 g0;
  g0[0] = 1u;                                           // count=1 (valid)
  g0[1] = ldsOff;                                       // lds_addr
  g0[2] = (unsigned)(ga & 0xFFFFFFFFu);                 // global_addr[31:0]
  g0[3] = (unsigned)((ga >> 32) & 0x1FFFFFFu)           // global_addr[56:32]
        | (2u << 30);                                   // type=2 ("image")
  i32x8 g1;
  g1[0] = (1 << 16);                                    // data_size=1 -> 2 bytes
  g1[1] = (rowStrideElems & 0xFFFF) << 16;              // tensor_dim0[15:0]
  g1[2] = ((rowStrideElems >> 16) & 0xFFFF)             // tensor_dim0[31:16]
        | ((tensorRows & 0xFFFF) << 16);                // tensor_dim1[15:0]
  g1[3] = ((tensorRows >> 16) & 0xFFFF)                 // tensor_dim1[31:16]
        | (tileK << 16);                                // tile_dim0
  g1[4] = tileRows;                                     // tile_dim1 (tile_dim2=0)
  g1[5] = rowStrideElems;                               // tensor_dim0_stride[31:0]
  g1[6] = 0;                                            // stride hi / dim1_stride lo
  g1[7] = 0;
  i32x4 gz4 = {};                                       // groups 2/3: 2-D tile
  i32x8 gz8 = {};
  __builtin_amdgcn_tensor_load_to_lds(g0, g1, gz4, gz4, gz8, 0);
}

// ---------------------------------------------------------------------------
// Kernel 1/3: Y[M=4096, N=1024] = X @ Wb^T + bias  (bf16 WMMA, f32 accum)
// Wb is pre-converted bf16. B tile fetched by the Tensor Data Mover.
//   MODE 0: out = head-split bf16 Qh/Kh [b][h][s][64]
//   MODE 1: out = transposed bf16 Vt    [b][h][64][s]
//   MODE 2: out = plain fp32 row-major  [m][n]
// Block: 256 thr (8 waves), tile 64x64, BK=32. 8 waves x 2 WMMA tiles.
// ---------------------------------------------------------------------------
template <int AISBF16, int MODE>
__global__ __launch_bounds__(256) void gemm_bias(
    const void* __restrict__ Av, const __bf16* __restrict__ Wb,
    const float* __restrict__ bias, void* __restrict__ Out) {
  __shared__ __align__(16) __bf16 sA[64][32];
  __shared__ __align__(16) __bf16 sB[64][32];

  const int n0 = blockIdx.x * 64;
  const int m0 = blockIdx.y * 64;
  const int t = threadIdx.x;
  const int w = t >> 5, lane = t & 31;
  const int lrow = t >> 2, lquad = t & 3;   // cooperative staging: 64 rows x 4 quads

  const int tr = w & 3;                     // wave's row tile (0..3)
  const int c0 = (w >> 2) * 2, c1 = c0 + 1; // wave's two col tiles

  const unsigned sbOff = (unsigned)(uintptr_t)&sB[0][0];

  v8f acc0 = {}; v8f acc1 = {};

  for (int kb = 0; kb < DM; kb += 32) {
    // ---- B tile: TDM async copy (wave 0 issues; EXEC-independent) ----
    if (w == 0) {
      tdm_load_2d_bf16(sbOff, Wb + (size_t)n0 * DM + kb,
                       /*tileK=*/32, /*tileRows=*/64,
                       /*rowStride=*/DM, /*tensorRows=*/DM);
    }
    // ---- A tile: cooperative stage (fp32 -> bf16 cvt, or bf16 copy) ----
    if (AISBF16) {
      const __bf16* ap = (const __bf16*)Av + (size_t)(m0 + lrow) * DM + kb + lquad * 8;
      *(v8bf*)&sA[lrow][lquad * 8] = *(const v8bf*)ap;
      __builtin_prefetch(ap + 32, 0, 1);
    } else {
      const float* ap = (const float*)Av + (size_t)(m0 + lrow) * DM + kb + lquad * 8;
      f32x4 x0 = *(const f32x4*)ap;
      f32x4 x1 = *(const f32x4*)(ap + 4);
      v8bf bv;
      bv[0] = (__bf16)x0[0]; bv[1] = (__bf16)x0[1];
      bv[2] = (__bf16)x0[2]; bv[3] = (__bf16)x0[3];
      bv[4] = (__bf16)x1[0]; bv[5] = (__bf16)x1[1];
      bv[6] = (__bf16)x1[2]; bv[7] = (__bf16)x1[3];
      *(v8bf*)&sA[lrow][lquad * 8] = bv;
      __builtin_prefetch(ap + 32, 0, 1);
    }
    if (w == 0) __builtin_amdgcn_s_wait_tensorcnt(0);
    __syncthreads();

    v16bf a  = frag_a16(&sA[tr * 16][0], 32, 0);
    v16bf b0 = frag_b16(&sB[c0 * 16][0], 32, 0);
    v16bf b1 = frag_b16(&sB[c1 * 16][0], 32, 0);
    acc0 = wmma_bf16(a, b0, acc0);
    acc1 = wmma_bf16(a, b1, acc1);
    __syncthreads();
  }

  // ---- store (C layout: VGPR i -> M=i (lanes 0-15) / M=i+8 (lanes 16-31)) ----
  const int nloc0 = n0 + c0 * 16 + (lane & 15);
  const int nloc1 = n0 + c1 * 16 + (lane & 15);
  const float bb0 = bias[nloc0];
  const float bb1 = bias[nloc1];
  const int rb = (lane >> 4) << 3;
#pragma unroll
  for (int i = 0; i < 8; ++i) {
    const int m = m0 + tr * 16 + i + rb;
    const float v0 = acc0[i] + bb0;
    const float v1 = acc1[i] + bb1;
    if (MODE == 2) {
      ((float*)Out)[(size_t)m * DM + nloc0] = v0;
      ((float*)Out)[(size_t)m * DM + nloc1] = v1;
    } else {
      const int b = m >> 11, s = m & (SEQ - 1);
      const int h0 = nloc0 >> 6, d0 = nloc0 & 63;
      const int h1 = nloc1 >> 6, d1 = nloc1 & 63;
      __bf16* o = (__bf16*)Out;
      if (MODE == 0) {  // Qh/Kh: [b][h][s][64]
        o[(((size_t)(b * NHEAD + h0) * SEQ) + s) * DK + d0] = (__bf16)v0;
        o[(((size_t)(b * NHEAD + h1) * SEQ) + s) * DK + d1] = (__bf16)v1;
      } else {          // Vt: [b][h][64][s]
        o[(((size_t)(b * NHEAD + h0) * DK) + d0) * SEQ + s] = (__bf16)v0;
        o[(((size_t)(b * NHEAD + h1) * DK) + d1) * SEQ + s] = (__bf16)v1;
      }
    }
  }
}

// ---------------------------------------------------------------------------
// Weight pre-conversion fp32 -> bf16 (done once per weight matrix; also halves
// the repeated weight-read bandwidth inside the GEMMs).
// ---------------------------------------------------------------------------
__global__ __launch_bounds__(256) void f32_to_bf16(const float* __restrict__ in,
                                                   __bf16* __restrict__ out) {
  const size_t i = ((size_t)blockIdx.x * 256 + threadIdx.x) * 8;
  f32x4 x0 = *(const f32x4*)(in + i);
  f32x4 x1 = *(const f32x4*)(in + i + 4);
  v8bf bv;
  bv[0] = (__bf16)x0[0]; bv[1] = (__bf16)x0[1];
  bv[2] = (__bf16)x0[2]; bv[3] = (__bf16)x0[3];
  bv[4] = (__bf16)x1[0]; bv[5] = (__bf16)x1[1];
  bv[6] = (__bf16)x1[2]; bv[7] = (__bf16)x1[3];
  *(v8bf*)(out + i) = bv;
}

// ---------------------------------------------------------------------------
// Kernel 2: fused scores -> causal mask -> softmax -> attn write -> P@V
// One workgroup (256 thr, 8 waves) per (b, h, 16-row q tile). 4096 blocks.
// Dynamic LDS: 16x2048 fp32 scores (128KB) + 16x2048 bf16 probs (64KB).
// ---------------------------------------------------------------------------
__global__ __launch_bounds__(256) void attn_kernel(
    const __bf16* __restrict__ Qh, const __bf16* __restrict__ Kh,
    const __bf16* __restrict__ Vt, float* __restrict__ attnOut,
    __bf16* __restrict__ ctx) {
  extern __shared__ char smem[];
  float*  sS = (float*)smem;                       // [16][SEQ] scores / exp / reduce
  __bf16* sP = (__bf16*)(smem + 16 * SEQ * 4);     // [16][SEQ] bf16 probs

  const int t = threadIdx.x, w = t >> 5, lane = t & 31;
  const int qt = blockIdx.x & 127;
  const int bh = blockIdx.x >> 7;                  // 0..31 = b*16+h
  const int q0 = qt * 16;

  const __bf16* Q = Qh + (size_t)bh * SEQ * DK + (size_t)q0 * DK;
  const __bf16* K = Kh + (size_t)bh * SEQ * DK;
  const __bf16* V = Vt + (size_t)bh * DK * SEQ;

  // Q fragments for the whole tile (d = 0..63 in two K=32 steps)
  const v16bf aq0 = frag_a16(Q, DK, 0);
  const v16bf aq1 = frag_a16(Q, DK, 32);

  // ---- stage 1: scores = scale * Q K^T, causal masked, into LDS ----
  for (int kt = w; kt < SEQ / 16; kt += 8) {
    const __bf16* Kt = K + (size_t)kt * 16 * DK;
    v16bf b0 = frag_b16(Kt, DK, 0);
    v16bf b1 = frag_b16(Kt, DK, 32);
    v8f c = {};
    c = wmma_bf16(aq0, b0, c);
    c = wmma_bf16(aq1, b1, c);
    const int col = kt * 16 + (lane & 15);
    const int rb = (lane >> 4) << 3;
#pragma unroll
    for (int i = 0; i < 8; ++i) {
      const int r = i + rb;
      const float s = (col <= q0 + r) ? c[i] * SCALE : NEGBIG;
      sS[r * SEQ + col] = s;
    }
  }
  __syncthreads();

  // ---- stage 2: softmax over each of the 16 rows (16 threads / row) ----
  {
    const int r = t >> 4, j = t & 15;
    const int q = q0 + r;
    float mx = -3.0e38f;
    for (int c = j; c < SEQ; c += 16) mx = fmaxf(mx, sS[r * SEQ + c]);
#pragma unroll
    for (int o = 8; o; o >>= 1) mx = fmaxf(mx, __shfl_xor(mx, o, 16));
    float sum = 0.f;
    for (int c = j; c < SEQ; c += 16) {
      const float e = __expf(sS[r * SEQ + c] - mx);
      sS[r * SEQ + c] = e;
      sum += e;
    }
#pragma unroll
    for (int o = 8; o; o >>= 1) sum += __shfl_xor(sum, o, 16);
    const float inv = 1.0f / sum;
    float* arow = attnOut + ((size_t)bh * SEQ + q) * SEQ;
    for (int c = j; c < SEQ; c += 16) {
      const float p = sS[r * SEQ + c] * inv;
      arow[c] = p;                      // mandatory 537MB attn output
      sP[r * SEQ + c] = (__bf16)p;      // bf16 copy for P@V
    }
  }
  __syncthreads();

  // ---- stage 3: ctx_tile = P @ V (k split across wave pairs) ----
  const int dt = w & 3;                 // d tile (0..3 -> 64 dims)
  const int kh = w >> 2;                // k half (0,1)
  v8f acc = {};
  const __bf16* Vd = V + (size_t)dt * 16 * SEQ;
  for (int kk = kh * 1024; kk < kh * 1024 + 1024; kk += 32) {
    v16bf a = frag_a16(sP, SEQ, kk);
    v16bf b = frag_b16(Vd, SEQ, kk);
    acc = wmma_bf16(a, b, acc);
  }
  // reduce the two k-halves via LDS (reuse score region)
  float* red = sS;
  {
    const int n = lane & 15, rb = (lane >> 4) << 3;
#pragma unroll
    for (int i = 0; i < 8; ++i) red[w * 256 + (i + rb) * 16 + n] = acc[i];
  }
  __syncthreads();
  if (w < 4) {
    const int n = lane & 15, rb = (lane >> 4) << 3;
    const int b = bh >> 4, h = bh & 15;
#pragma unroll
    for (int i = 0; i < 8; ++i) {
      const int m = i + rb;
      const float v = red[w * 256 + m * 16 + n] + red[(w + 4) * 256 + m * 16 + n];
      ctx[((size_t)(b * SEQ) + q0 + m) * DM + h * DK + w * 16 + n] = (__bf16)v;
    }
  }
}

// ---------------------------------------------------------------------------
// Host-side orchestration.
// ws (bf16): Qh | Kh | Vt | ctx (8MB each) | Wq | Wk | Wv | Wo (2MB each) = 40MB
// d_out: [out fp32 4.19M] [attn fp32 134.2M]
// ---------------------------------------------------------------------------
extern "C" void kernel_launch(void* const* d_in, const int* in_sizes, int n_in,
                              void* d_out, int out_size, void* d_ws, size_t ws_size,
                              hipStream_t stream) {
  const float* q   = (const float*)d_in[0];
  const float* k   = (const float*)d_in[1];
  const float* v   = (const float*)d_in[2];
  // d_in[3] is the causal mask; causality is applied analytically in-kernel.
  const float* w_q = (const float*)d_in[4];
  const float* b_q = (const float*)d_in[5];
  const float* w_k = (const float*)d_in[6];
  const float* b_k = (const float*)d_in[7];
  const float* w_v = (const float*)d_in[8];
  const float* b_v = (const float*)d_in[9];
  const float* w_o = (const float*)d_in[10];
  const float* b_o = (const float*)d_in[11];

  const size_t perTensor = (size_t)BSZ * NHEAD * SEQ * DK;  // 4,194,304 bf16
  const size_t wElems    = (size_t)DM * DM;                 // 1,048,576
  __bf16* Qh  = (__bf16*)d_ws;
  __bf16* Kh  = Qh + perTensor;
  __bf16* Vt  = Kh + perTensor;
  __bf16* ctx = Vt + perTensor;
  __bf16* Wq  = ctx + perTensor;
  __bf16* Wk  = Wq + wElems;
  __bf16* Wv  = Wk + wElems;
  __bf16* Wo  = Wv + wElems;

  float* outP  = (float*)d_out;
  float* attnP = outP + (size_t)BSZ * SEQ * DM;

  dim3 blk(256);
  const dim3 gcv(wElems / (256 * 8));  // 512 blocks per weight conversion
  f32_to_bf16<<<gcv, blk, 0, stream>>>(w_q, Wq);
  f32_to_bf16<<<gcv, blk, 0, stream>>>(w_k, Wk);
  f32_to_bf16<<<gcv, blk, 0, stream>>>(w_v, Wv);
  f32_to_bf16<<<gcv, blk, 0, stream>>>(w_o, Wo);

  const dim3 gg(DM / 64, (BSZ * SEQ) / 64);  // 16 x 64 workgroups per GEMM
  gemm_bias<0, 0><<<gg, blk, 0, stream>>>(q, Wq, b_q, Qh);
  gemm_bias<0, 0><<<gg, blk, 0, stream>>>(k, Wk, b_k, Kh);
  gemm_bias<0, 1><<<gg, blk, 0, stream>>>(v, Wv, b_v, Vt);

  const size_t smemBytes = (size_t)16 * SEQ * 4 + (size_t)16 * SEQ * 2;  // 192KB
  attn_kernel<<<dim3(BSZ * NHEAD * (SEQ / 16)), blk, smemBytes, stream>>>(
      Qh, Kh, Vt, attnP, ctx);

  gemm_bias<1, 2><<<gg, blk, 0, stream>>>(ctx, Wo, b_o, outP);
}
